// MultiHeadSelfAttention_89266600280716
// MI455X (gfx1250) — matrix operand order repaired and checked by reading, not compiled, not run on hardware
//
#include <hip/hip_runtime.h>
#include <hip/hip_bf16.h>

// ---------------------------------------------------------------------------
// MHA forward for MI455X (gfx1250, wave32, WMMA + async LDS staging).
// Pipeline: cast -> QKV GEMM (bf16 WMMA, LDS double-buffered B panel)
//           -> scores+exp (WMMA + v_exp_f32 + shfl_xor rowsum)
//           -> normalize -> PV GEMM (WMMA) -> O-proj GEMM (WMMA, LDS).
// ---------------------------------------------------------------------------

typedef __bf16 bf16;
typedef __attribute__((ext_vector_type(16))) __bf16 v16bf;
typedef __attribute__((ext_vector_type(8)))  float   v8f;
typedef int v4i __attribute__((vector_size(16)));

#define B_   2
#define T_   2048
#define D_   1024
#define H_   16
#define DH_  64
#define BT_  4096          // B*T
#define D3_  3072          // 3*D
#define BH_  32            // B*H

// LDS B-panel layout: 64 columns x 32 K-values, padded to 48 bf16 (96B) per
// column so ds_load_b128 stays 16B-aligned; two buffers for double buffering.
#define LDS_COL_STRIDE 48
#define LDS_BUF_ELEMS  (64 * LDS_COL_STRIDE)     // 3072 bf16 = 6KB per buffer

// ---------------- CDNA5 async copy (ASYNCcnt path) --------------------------
__device__ __forceinline__ void async_copy16(const bf16* g, bf16* l) {
#if defined(__has_builtin) && __has_builtin(__builtin_amdgcn_global_load_async_to_lds_b128)
    __builtin_amdgcn_global_load_async_to_lds_b128((v4i*)g, (v4i*)l, 0, 0);
#else
    *(uint4*)l = *(const uint4*)g;               // sync fallback (ds_store_b128)
#endif
}

__device__ __forceinline__ void wait_async_all() {
#if defined(__has_builtin) && __has_builtin(__builtin_amdgcn_s_wait_asynccnt)
    __builtin_amdgcn_s_wait_asynccnt(0);
#else
    asm volatile("s_wait_asynccnt 0" ::: "memory");
#endif
}

// ---------------- fragment loaders (contiguous k order, k = 16*(lane/16) + s) --------------------
// A-matrix 16x32 bf16, row-major source, stride lda (elements).
__device__ __forceinline__ v16bf load_a_bf16(const bf16* __restrict__ base, int lda,
                                             int m0, int k0, int lane) {
    int row = m0 + (lane & 15);
    int kh  = lane >> 4;
    const bf16* p = base + (size_t)row * lda + k0 + kh * 16;
    const bf16* q = p + 8;
    v16bf f;
#pragma unroll
    for (int i = 0; i < 8; ++i) { f[i] = p[i]; f[8 + i] = q[i]; }
    return f;
}

// Same A layout, f32 source converted to bf16 in-register (for P tiles).
__device__ __forceinline__ v16bf load_a_f32(const float* __restrict__ base, int lda,
                                            int m0, int k0, int lane) {
    int row = m0 + (lane & 15);
    int kh  = lane >> 4;
    const float* p = base + (size_t)row * lda + k0 + kh * 16;
    const float* q = p + 8;
    v16bf f;
#pragma unroll
    for (int i = 0; i < 8; ++i) { f[i] = (bf16)p[i]; f[8 + i] = (bf16)q[i]; }
    return f;
}

// B-matrix 32x16 bf16 from [N,K]-major global memory: 16 contiguous per lane.
__device__ __forceinline__ v16bf load_b_bf16(const bf16* __restrict__ base, int ldb,
                                             int n0, int k0, int lane) {
    int col = n0 + (lane & 15);
    const bf16* p = base + (size_t)col * ldb + k0 + (lane >> 4) * 16;
    v16bf f;
#pragma unroll
    for (int i = 0; i < 16; ++i) f[i] = p[i];
    return f;
}

// B fragment from the LDS staging panel (col-strided, padded).
__device__ __forceinline__ v16bf load_b_lds(const bf16* sb, int nsub, int lane) {
    const bf16* p = sb + (nsub + (lane & 15)) * LDS_COL_STRIDE + (lane >> 4) * 16;
    v16bf f;
#pragma unroll
    for (int i = 0; i < 16; ++i) f[i] = p[i];
    return f;
}

// ---------------- shared GEMM core: 16(M) x 64(N) per wave ------------------
// A bf16 [M,K] row-major; WT bf16 [N,K]; B panel (64 cols) staged in LDS with
// async copies, double buffered; each wave reuses its A frag across 4 WMMAs.
struct Acc4 { v8f c[4]; };

__device__ __forceinline__ Acc4 gemm_core(const bf16* __restrict__ A,
                                          const bf16* __restrict__ WT,
                                          int Kdim, int m0, int n0, bf16* sh) {
    const int lane = threadIdx.x & 31;
    const int tid  = threadIdx.x;
    const int ccol = tid >> 2;                   // 0..63: column this thread copies
    const int cchk = tid & 3;                    // 4 x 16B chunks cover 32 bf16
    const bf16* gsrc = WT + (size_t)(n0 + ccol) * Kdim + cchk * 8;
    bf16* ldst = sh + ccol * LDS_COL_STRIDE + cchk * 8;

    Acc4 acc;
#pragma unroll
    for (int f = 0; f < 4; ++f) acc.c[f] = v8f{};

    // prologue: stage k0 = 0 panel into buffer 0
    async_copy16(gsrc, ldst);
    wait_async_all();
    __syncthreads();

    for (int k0 = 0; k0 < Kdim; k0 += 32) {
        int cur = (k0 >> 5) & 1;
        if (k0 + 32 < Kdim)                      // prefetch next panel async
            async_copy16(gsrc + (k0 + 32), ldst + (cur ^ 1) * LDS_BUF_ELEMS);

        v16bf a = load_a_bf16(A, Kdim, m0, k0, lane);
        const bf16* sb = sh + cur * LDS_BUF_ELEMS;
#pragma unroll
        for (int f = 0; f < 4; ++f) {
            v16bf b = load_b_lds(sb, f * 16, lane);
            acc.c[f] = __builtin_amdgcn_wmma_f32_16x16x32_bf16(false, a, false, b,
                                                               (short)0, acc.c[f],
                                                               false, false);
        }
        wait_async_all();
        __syncthreads();
    }
    return acc;
}

// ---------------- elementwise prep kernels ---------------------------------
__global__ void zero_f32_kernel(float* __restrict__ p, int n) {
    int i = blockIdx.x * blockDim.x + threadIdx.x;
    if (i < n) p[i] = 0.f;
}

__global__ void cvt_f32_to_bf16_kernel(const float* __restrict__ src,
                                       bf16* __restrict__ dst, int n) {
    int i = blockIdx.x * blockDim.x + threadIdx.x;
    if (i < n) dst[i] = (bf16)src[i];
}

// src [K,N] row-major f32 -> dst [N,K] bf16 (transpose + cast)
__global__ void transpose_cvt_kernel(const float* __restrict__ src,
                                     bf16* __restrict__ dst, int K, int N) {
    int i = blockIdx.x * blockDim.x + threadIdx.x;
    if (i >= K * N) return;
    int k = i / N, n = i % N;
    dst[(size_t)n * K + k] = (bf16)src[i];
}

// ---------------- GEMM 1: QKV = X @ Wqkv ------------------------------------
// Grid: (BT/128) x (3D/64) = 32*48 blocks, 8 waves each (one 16-row M tile).
__global__ void qkv_gemm_kernel(const bf16* __restrict__ X, const bf16* __restrict__ WT,
                                bf16* __restrict__ Q, bf16* __restrict__ K,
                                bf16* __restrict__ VT) {
    __shared__ bf16 sh[2 * LDS_BUF_ELEMS];
    const int lane = threadIdx.x & 31;
    const int wave = threadIdx.x >> 5;
    int bm = blockIdx.x / (D3_ / 64);
    int bn = blockIdx.x % (D3_ / 64);
    int m0 = bm * 128 + wave * 16;
    int n0 = bn * 64;

    Acc4 acc = gemm_core(X, WT, D_, m0, n0, sh);

    int rbase = (lane >> 4) * 8;
#pragma unroll
    for (int f = 0; f < 4; ++f) {
        int col = n0 + f * 16 + (lane & 15);
#pragma unroll
        for (int r = 0; r < 8; ++r) {
            int row = m0 + rbase + r;            // token index
            int bb = row / T_, t = row % T_;
            bf16 v = (bf16)acc.c[f][r];
            if (col < D_) {
                int h = col / DH_, d = col % DH_;
                Q[(((size_t)(bb * H_ + h)) * T_ + t) * DH_ + d] = v;
            } else if (col < 2 * D_) {
                int cc = col - D_, h = cc / DH_, d = cc % DH_;
                K[(((size_t)(bb * H_ + h)) * T_ + t) * DH_ + d] = v;
            } else {
                int cc = col - 2 * D_, h = cc / DH_, d = cc % DH_;
                VT[(((size_t)(bb * H_ + h)) * DH_ + d) * T_ + t] = v;
            }
        }
    }
}

// ---------------- scores: exp((Q K^T)/8) with causal mask -------------------
__global__ void scores_kernel(const bf16* __restrict__ Q, const bf16* __restrict__ K,
                              float* __restrict__ Wout, float* __restrict__ rowsum) {
    const int lane = threadIdx.x & 31;
    const int wave = threadIdx.x >> 5;
    const int ntk = T_ / 16;                      // 128
    int tile = blockIdx.x * 8 + wave;             // grid exactly covers all tiles
    int bh  = tile / (ntk * ntk);
    int rem = tile % (ntk * ntk);
    int qt = rem / ntk, kt = rem % ntk;

    float* Wbh  = Wout + (size_t)bh * T_ * T_;
    int col   = kt * 16 + (lane & 15);
    int rbase = (lane >> 4) * 8;

    if (kt > qt) {                                // fully masked tile -> zeros
#pragma unroll
        for (int r = 0; r < 8; ++r)
            Wbh[(size_t)(qt * 16 + rbase + r) * T_ + col] = 0.f;
        return;
    }

    const bf16* Qb = Q + (size_t)bh * T_ * DH_;
    const bf16* Kb = K + (size_t)bh * T_ * DH_;
    v8f c = {};
#pragma unroll
    for (int k0 = 0; k0 < DH_; k0 += 32) {
        v16bf a = load_a_bf16(Qb, DH_, qt * 16, k0, lane);
        v16bf b = load_b_bf16(Kb, DH_, kt * 16, k0, lane);
        c = __builtin_amdgcn_wmma_f32_16x16x32_bf16(false, a, false, b,
                                                    (short)0, c, false, false);
    }

    float* rs = rowsum + (size_t)bh * T_;
#pragma unroll
    for (int r = 0; r < 8; ++r) {
        int q = qt * 16 + rbase + r;
        float e = (col > q) ? 0.f : __expf(c[r] * 0.125f);
        Wbh[(size_t)q * T_ + col] = e;
        // lanes 0-15 share one row, lanes 16-31 another: butterfly within halves
        float s = e;
        s += __shfl_xor(s, 1, 32);
        s += __shfl_xor(s, 2, 32);
        s += __shfl_xor(s, 4, 32);
        s += __shfl_xor(s, 8, 32);
        if ((lane & 15) == 0) atomicAdd(&rs[q], s);
    }
}

// ---------------- normalize weights in place --------------------------------
__global__ void norm_kernel(float* __restrict__ W, const float* __restrict__ rowsum) {
    size_t i = ((size_t)blockIdx.x * blockDim.x + threadIdx.x) * 4;
    size_t row = i >> 11;                         // / T_
    float inv = 1.f / rowsum[row];
    float4 w = *(float4*)(W + i);
    w.x *= inv; w.y *= inv; w.z *= inv; w.w *= inv;
    *(float4*)(W + i) = w;
}

// ---------------- GEMM 2: O = P @ V -----------------------------------------
// Streams 537MB of P once: HBM-bound, so no LDS staging; causal-truncated K.
__global__ void pv_gemm_kernel(const float* __restrict__ P, const bf16* __restrict__ VT,
                               bf16* __restrict__ AO) {
    const int lane = threadIdx.x & 31;
    const int wave = threadIdx.x >> 5;
    int tile = blockIdx.x * 8 + wave;             // bh(32) x qt(128) x dt(4)
    int dt = tile & 3;
    int qt = (tile >> 2) & 127;
    int bh = tile >> 9;

    const float* Pb = P  + (size_t)bh * T_ * T_;
    const bf16*  Vb = VT + (size_t)bh * DH_ * T_;

    v8f c = {};
    int kmax = (qt + 1) * 16;                     // causal: tail weights are 0
    for (int k0 = 0; k0 < kmax; k0 += 32) {
        v16bf a = load_a_f32(Pb, T_, qt * 16, k0, lane);
        v16bf b = load_b_bf16(Vb, T_, dt * 16, k0, lane);
        c = __builtin_amdgcn_wmma_f32_16x16x32_bf16(false, a, false, b,
                                                    (short)0, c, false, false);
    }

    int bb = bh >> 4, h = bh & 15;
    int col   = dt * 16 + (lane & 15);
    int rbase = (lane >> 4) * 8;
#pragma unroll
    for (int r = 0; r < 8; ++r) {
        int q = qt * 16 + rbase + r;
        AO[((size_t)(bb * T_ + q)) * D_ + h * DH_ + col] = (bf16)c[r];
    }
}

// ---------------- GEMM 3: out = AO @ Wo + bo --------------------------------
__global__ void oproj_gemm_kernel(const bf16* __restrict__ AO, const bf16* __restrict__ WoT,
                                  const float* __restrict__ bo, float* __restrict__ Out) {
    __shared__ bf16 sh[2 * LDS_BUF_ELEMS];
    const int lane = threadIdx.x & 31;
    const int wave = threadIdx.x >> 5;
    int bm = blockIdx.x / (D_ / 64);
    int bn = blockIdx.x % (D_ / 64);
    int m0 = bm * 128 + wave * 16;
    int n0 = bn * 64;

    Acc4 acc = gemm_core(AO, WoT, D_, m0, n0, sh);

    int rbase = (lane >> 4) * 8;
#pragma unroll
    for (int f = 0; f < 4; ++f) {
        int col = n0 + f * 16 + (lane & 15);
        float bias = bo[col];
#pragma unroll
        for (int r = 0; r < 8; ++r)
            Out[(size_t)(m0 + rbase + r) * D_ + col] = acc.c[f][r] + bias;
    }
}

// ---------------- launcher ---------------------------------------------------
extern "C" void kernel_launch(void* const* d_in, const int* in_sizes, int n_in,
                              void* d_out, int out_size, void* d_ws, size_t ws_size,
                              hipStream_t stream) {
    const float* x    = (const float*)d_in[0];
    // d_in[1] = attn_mask (causal, known at compile time -> unused)
    const float* Wqkv = (const float*)d_in[2];
    const float* Wo   = (const float*)d_in[3];
    const float* bo   = (const float*)d_in[4];

    float* out_o = (float*)d_out;                           // [B,T,D]
    float* out_w = (float*)d_out + (size_t)BT_ * D_;        // [B,H,T,T]

    char* ws = (char*)d_ws;
    const size_t OFF_XB   = 0;
    const size_t OFF_WQKT = OFF_XB   + (size_t)BT_ * D_  * 2;
    const size_t OFF_WOT  = OFF_WQKT + (size_t)D_  * D3_ * 2;
    const size_t OFF_Q    = OFF_WOT  + (size_t)D_  * D_  * 2;
    const size_t OFF_K    = OFF_Q    + (size_t)BH_ * T_ * DH_ * 2;
    const size_t OFF_VT   = OFF_K    + (size_t)BH_ * T_ * DH_ * 2;
    const size_t OFF_RS   = OFF_VT   + (size_t)BH_ * T_ * DH_ * 2;
    const size_t OFF_AO   = OFF_RS   + (size_t)BH_ * T_ * 4;

    bf16*  xb     = (bf16*)(ws + OFF_XB);
    bf16*  wqkvT  = (bf16*)(ws + OFF_WQKT);
    bf16*  woT    = (bf16*)(ws + OFF_WOT);
    bf16*  Qb     = (bf16*)(ws + OFF_Q);
    bf16*  Kb     = (bf16*)(ws + OFF_K);
    bf16*  VTb    = (bf16*)(ws + OFF_VT);
    float* rowsum = (float*)(ws + OFF_RS);
    bf16*  AO     = (bf16*)(ws + OFF_AO);

    // 1) zero row sums (atomically accumulated each call)
    zero_f32_kernel<<<(BH_ * T_ + 255) / 256, 256, 0, stream>>>(rowsum, BH_ * T_);
    // 2) casts / transposes
    cvt_f32_to_bf16_kernel<<<(BT_ * D_) / 256, 256, 0, stream>>>(x, xb, BT_ * D_);
    transpose_cvt_kernel<<<(D_ * D3_) / 256, 256, 0, stream>>>(Wqkv, wqkvT, D_, D3_);
    transpose_cvt_kernel<<<(D_ * D_) / 256, 256, 0, stream>>>(Wo, woT, D_, D_);
    // 3) QKV projection: 32 x 48 blocks of 128x64
    qkv_gemm_kernel<<<32 * 48, 256, 0, stream>>>(xb, wqkvT, Qb, Kb, VTb);
    // 4) scores + exp: 32 * 128 * 128 = 524288 tiles, 8 waves/block
    scores_kernel<<<524288 / 8, 256, 0, stream>>>(Qb, Kb, out_w, rowsum);
    // 5) softmax normalization (134217728 elems, 4/thread)
    norm_kernel<<<(size_t)(134217728 / 4) / 256, 256, 0, stream>>>(out_w, rowsum);
    // 6) P @ V: 32 * 128 * 4 = 16384 tiles
    pv_gemm_kernel<<<16384 / 8, 256, 0, stream>>>(out_w, VTb, AO);
    // 7) output projection: 32 x 16 blocks of 128x64
    oproj_gemm_kernel<<<32 * 16, 256, 0, stream>>>(AO, woT, bo, out_o);
}
